// EchoNetConvMamba_12335146074891
// MI455X (gfx1250) — compile-verified, matrix operand order checked by reading, not run on hardware
//
#include <hip/hip_runtime.h>

typedef _Float16 half_t;
typedef _Float16 v8h  __attribute__((ext_vector_type(8)));
typedef _Float16 v16h __attribute__((ext_vector_type(16)));
typedef float    v8f  __attribute__((ext_vector_type(8)));

static inline int ceil_div(int a, int b) { return (a + b - 1) / b; }

// ---------------- device helpers ----------------

static __device__ __forceinline__ float act_f(float v, int act) {
  switch (act) {
    case 1: return 0.5f * v * (1.f + erff(v * 0.70710678118f)); // exact GELU
    case 2: return v / (1.f + __expf(-v));                      // SiLU
    case 3: return v > 0.f ? v : 0.f;                           // ReLU
    case 4: return (v > 20.f) ? v : log1pf(__expf(v));          // softplus
    case 5: return 1.f / (1.f + __expf(-v));                    // sigmoid
    default: return v;
  }
}

// async global->LDS copy of 16B (GLOBAL_LOAD_ASYNC_TO_LDS_B128, ASYNCcnt).
// The LDS base pointer is passed as an extra (unreferenced) operand so the
// shared array's address ESCAPES into the asm: with the "memory" clobber the
// compiler must then treat each async op as a potential writer of the LDS
// object, keeping the later ds_load fragment reads alive.
static __device__ __forceinline__ void async_ld_b128(unsigned lds_byte_addr,
                                                     const void* gptr,
                                                     void* lds_base) {
  asm volatile("global_load_async_to_lds_b128 %0, %1, off"
               :: "v"(lds_byte_addr), "v"(gptr), "v"(lds_base)
               : "memory");
}
static __device__ __forceinline__ void wait_async_le4() {
  asm volatile("s_wait_asynccnt 0x4" ::: "memory");
}
static __device__ __forceinline__ void wait_async_0() {
  asm volatile("s_wait_asynccnt 0x0" ::: "memory");
}

// ---------------- WMMA GEMM: C[M,N] = act(A[M,K] * W[N,K]^T + bias) ----------------
// Invariants guaranteed by the host code (no guards on the load path):
//   - M is a multiple of 128
//   - K is a multiple of 32 (dt_proj zero-padded 48 -> 64)
//   - A/W workspace buffers cover ceil(N/128)*128 rows (over-read rows only
//     feed out-of-range output columns, guarded at the store)
// 128x128 block tile, 256 threads = 8 wave32s in a 4(M) x 2(N) grid; each wave
// computes a 32x64 slab = 2x4 WMMA tiles. LDS tiles are kept in WMMA fragment
// order (ISA 7.12.2): a lane's v16h operand is two contiguous 8-half runs
// along K, so staging is pure 16B moves — done with double-buffered
// GLOBAL_LOAD_ASYNC_TO_LDS_B128 (no VGPR round trip, ASYNCcnt pipelined).

__global__ __launch_bounds__(256)
void k_gemm_wmma(const half_t* __restrict__ A, const half_t* __restrict__ W,
                 const float* __restrict__ bias, float* __restrict__ Cf,
                 half_t* __restrict__ Ch, int M, int N, int K,
                 int act, int bshift, int outh)
{
  // [buf][ A: halfs 0..4095 | B: halfs 4096..8191 ]  (32 KB total)
  __shared__ __attribute__((aligned(32))) half_t lds[2][8192];
  void* ldsp = (void*)lds;
  const int tid  = threadIdx.x;
  const int bm   = blockIdx.y * 128;
  const int bn   = blockIdx.x * 128;
  const int lane = tid & 31;
  const int wave = tid >> 5;
  const int wm   = wave >> 1;   // 0..3 -> m-blocks 2*wm, 2*wm+1
  const int wn   = wave & 1;    // 0..1 -> n-blocks 4*wn .. 4*wn+3

  v8f acc[2][4];
  for (int i = 0; i < 2; ++i)
    for (int j = 0; j < 4; ++j)
      for (int r = 0; r < 8; ++r) acc[i][j][r] = 0.f;

  // each thread moves 2 16B chunks of A and 2 of B per K step
  // chunk q (0..511): blk = q>>6, ln = (q&63)>>1, sel = q&1
  const int q0   = tid;         // chunks tid and tid+256
  const int q1   = tid + 256;
  const int blk0 = q0 >> 6, ln0 = (q0 & 63) >> 1, sel0 = q0 & 1;
  const int blk1 = q1 >> 6, ln1 = (q1 & 63) >> 1, sel1 = q1 & 1;
  const int mrow0 = bm + blk0 * 16 + (ln0 & 15);
  const int mrow1 = bm + blk1 * 16 + (ln1 & 15);
  const int nrow0 = bn + blk0 * 16 + (ln0 & 15);
  const int nrow1 = bn + blk1 * 16 + (ln1 & 15);
  const int koff0 = sel0 * 16 + ((ln0 >= 16) ? 8 : 0);
  const int koff1 = sel1 * 16 + ((ln1 >= 16) ? 8 : 0);

  const half_t* gA0 = A + (size_t)mrow0 * K + koff0;
  const half_t* gA1 = A + (size_t)mrow1 * K + koff1;
  const half_t* gB0 = W + (size_t)nrow0 * K + koff0;
  const half_t* gB1 = W + (size_t)nrow1 * K + koff1;
  const unsigned la0 = (unsigned)q0 * 16u;          // byte offset of A chunk
  const unsigned la1 = (unsigned)q1 * 16u;
  const unsigned BUFB = 8192u;                      // B region byte offset
  const unsigned BUFS = 16384u;                     // ping/pong stride (bytes)

  // prologue: stage tile 0 into buffer 0
  async_ld_b128(la0,        gA0, ldsp);
  async_ld_b128(la1,        gA1, ldsp);
  async_ld_b128(BUFB + la0, gB0, ldsp);
  async_ld_b128(BUFB + la1, gB1, ldsp);

  int buf = 0;
  for (int k0 = 0; k0 < K; k0 += 32) {
    // warm L2 two tiles ahead (speculative; OOB silently dropped)
    __builtin_prefetch(gA0 + k0 + 64, 0, 1);
    __builtin_prefetch(gB0 + k0 + 64, 0, 1);

    if (k0 + 32 < K) {
      const unsigned nb = (unsigned)(buf ^ 1) * BUFS;
      async_ld_b128(nb + la0,        gA0 + k0 + 32, ldsp);
      async_ld_b128(nb + la1,        gA1 + k0 + 32, ldsp);
      async_ld_b128(nb + BUFB + la0, gB0 + k0 + 32, ldsp);
      async_ld_b128(nb + BUFB + la1, gB1 + k0 + 32, ldsp);
      wait_async_le4();   // tile t's 4 ops retired (async loads complete in order)
    } else {
      wait_async_0();
    }
    __syncthreads();

    const half_t* LA = &lds[buf][0];
    const half_t* LB = &lds[buf][4096];
    v16h af0 = *(const v16h*)&LA[(wm * 2 + 0) * 512 + lane * 16];
    v16h af1 = *(const v16h*)&LA[(wm * 2 + 1) * 512 + lane * 16];
    v16h bf0 = *(const v16h*)&LB[(wn * 4 + 0) * 512 + lane * 16];
    v16h bf1 = *(const v16h*)&LB[(wn * 4 + 1) * 512 + lane * 16];
    v16h bf2 = *(const v16h*)&LB[(wn * 4 + 2) * 512 + lane * 16];
    v16h bf3 = *(const v16h*)&LB[(wn * 4 + 3) * 512 + lane * 16];

    acc[0][0] = __builtin_amdgcn_wmma_f32_16x16x32_f16(false, af0, false, bf0, (short)0, acc[0][0], false, false);
    acc[0][1] = __builtin_amdgcn_wmma_f32_16x16x32_f16(false, af0, false, bf1, (short)0, acc[0][1], false, false);
    acc[0][2] = __builtin_amdgcn_wmma_f32_16x16x32_f16(false, af0, false, bf2, (short)0, acc[0][2], false, false);
    acc[0][3] = __builtin_amdgcn_wmma_f32_16x16x32_f16(false, af0, false, bf3, (short)0, acc[0][3], false, false);
    acc[1][0] = __builtin_amdgcn_wmma_f32_16x16x32_f16(false, af1, false, bf0, (short)0, acc[1][0], false, false);
    acc[1][1] = __builtin_amdgcn_wmma_f32_16x16x32_f16(false, af1, false, bf1, (short)0, acc[1][1], false, false);
    acc[1][2] = __builtin_amdgcn_wmma_f32_16x16x32_f16(false, af1, false, bf2, (short)0, acc[1][2], false, false);
    acc[1][3] = __builtin_amdgcn_wmma_f32_16x16x32_f16(false, af1, false, bf3, (short)0, acc[1][3], false, false);

    __syncthreads();  // all waves done reading buf before it is overwritten
    buf ^= 1;
  }

  // C layout: VGPR r, lanes 0-15 -> M=r, lanes 16-31 -> M=r+8; N = lane&15
  const int mbase = bm + wm * 32 + ((lane >= 16) ? 8 : 0);
  const int nbase = bn + wn * 64 + (lane & 15);
  for (int i = 0; i < 2; ++i) {
    for (int j = 0; j < 4; ++j) {
      int n = nbase + j * 16;
      if (n >= N) continue;
      float bv = bias ? bias[n >> bshift] : 0.f;
      for (int r = 0; r < 8; ++r) {
        int m = mbase + i * 16 + r;
        if (m >= M) continue;
        float v = act_f(acc[i][j][r] + bv, act);
        if (outh) Ch[(size_t)m * N + n] = (half_t)v;
        else      Cf[(size_t)m * N + n] = v;
      }
    }
  }
}

// ---------------- elementwise / small kernels ----------------

__global__ void k_f2h(const float* __restrict__ s, half_t* __restrict__ d, int n) {
  int i = blockIdx.x * blockDim.x + threadIdx.x;
  if (i < n) d[i] = (half_t)s[i];
}

// stem: x(128,3,112,112) -> out(128,96,28,28), conv4x4 s4 + bias
__global__ void k_stem(const float* __restrict__ x, const float* __restrict__ w,
                       const float* __restrict__ b, float* __restrict__ out) {
  int i = blockIdx.x * blockDim.x + threadIdx.x;
  const int total = 128 * 96 * 28 * 28;
  if (i >= total) return;
  int ox = i % 28; int t = i / 28;
  int oy = t % 28; t /= 28;
  int c  = t % 96; int n = t / 96;
  float acc = b[c];
  const float* xb = x + (size_t)n * 3 * 12544;
  const float* wb = w + (size_t)c * 48;
  for (int ci = 0; ci < 3; ++ci)
    for (int ky = 0; ky < 4; ++ky)
      for (int kx = 0; kx < 4; ++kx)
        acc += xb[(size_t)ci * 12544 + (oy * 4 + ky) * 112 + (ox * 4 + kx)]
             * wb[ci * 16 + ky * 4 + kx];
  out[i] = acc;
}

// LayerNorm over channel dim of NCHW, in place
__global__ void k_ln2d(float* __restrict__ x, const float* __restrict__ g,
                       const float* __restrict__ b, int N, int C, int HW) {
  int i = blockIdx.x * blockDim.x + threadIdx.x;
  if (i >= N * HW) return;
  int n = i / HW, p = i % HW;
  float* base = x + (size_t)n * C * HW + p;
  float m = 0.f;
  for (int c = 0; c < C; ++c) m += base[(size_t)c * HW];
  m /= (float)C;
  float v = 0.f;
  for (int c = 0; c < C; ++c) { float d = base[(size_t)c * HW] - m; v += d * d; }
  v /= (float)C;
  float inv = rsqrtf(v + 1e-6f);
  for (int c = 0; c < C; ++c)
    base[(size_t)c * HW] = (base[(size_t)c * HW] - m) * inv * g[c] + b[c];
}

// LayerNorm over channels, NCHW -> f16 rows[(n*HW+p)*C + c]  (GEMM A)
__global__ void k_ln_rows_f16(const float* __restrict__ x, const float* __restrict__ g,
                              const float* __restrict__ b, half_t* __restrict__ rows,
                              int N, int C, int HW) {
  int i = blockIdx.x * blockDim.x + threadIdx.x;
  if (i >= N * HW) return;
  int n = i / HW, p = i % HW;
  const float* base = x + (size_t)n * C * HW + p;
  float m = 0.f;
  for (int c = 0; c < C; ++c) m += base[(size_t)c * HW];
  m /= (float)C;
  float v = 0.f;
  for (int c = 0; c < C; ++c) { float d = base[(size_t)c * HW] - m; v += d * d; }
  v /= (float)C;
  float inv = rsqrtf(v + 1e-6f);
  half_t* r = rows + (size_t)i * C;
  for (int c = 0; c < C; ++c)
    r[c] = (half_t)((base[(size_t)c * HW] - m) * inv * g[c] + b[c]);
}

// NCHW f32 -> f16 rows[(f*HW+p)*C + c] (GEMM A, no norm)
__global__ void k_nchw_rows_f16(const float* __restrict__ in, half_t* __restrict__ rows,
                                int F, int C, int HW) {
  int i = blockIdx.x * blockDim.x + threadIdx.x;
  int total = F * C * HW;
  if (i >= total) return;
  int p = i % HW; int t = i / HW;
  int c = t % C;  int f = t / C;
  rows[((size_t)(f * HW + p)) * C + c] = (half_t)in[i];
}

// depthwise 7x7, pad 3
__global__ void k_dwconv7(const float* __restrict__ in, const float* __restrict__ w,
                          const float* __restrict__ b, float* __restrict__ out,
                          int N, int C, int H) {
  int i = blockIdx.x * blockDim.x + threadIdx.x;
  int total = N * C * H * H;
  if (i >= total) return;
  int ox = i % H; int t = i / H;
  int oy = t % H; t /= H;
  int c  = t % C; int n = t / C;
  const float* ib = in + ((size_t)n * C + c) * H * H;
  const float* wb = w + (size_t)c * 49;
  float acc = b[c];
  for (int ky = 0; ky < 7; ++ky) {
    int y = oy + ky - 3; if (y < 0 || y >= H) continue;
    for (int kx = 0; kx < 7; ++kx) {
      int xx = ox + kx - 3; if (xx < 0 || xx >= H) continue;
      acc += ib[y * H + xx] * wb[ky * 7 + kx];
    }
  }
  out[i] = acc;
}

// cur += gamma[c] * rows32[(n*HW+p)*C + c]
__global__ void k_residual(float* __restrict__ cur, const float* __restrict__ rows,
                           const float* __restrict__ gamma, int N, int C, int HW) {
  int i = blockIdx.x * blockDim.x + threadIdx.x;
  int total = N * C * HW;
  if (i >= total) return;
  int p = i % HW; int t = i / HW;
  int c = t % C;  int n = t / C;
  cur[i] += gamma[c] * rows[((size_t)(n * HW + p)) * C + c];
}

// downsample conv 2x2 stride 2
__global__ void k_down2(const float* __restrict__ in, const float* __restrict__ w,
                        const float* __restrict__ b, float* __restrict__ out,
                        int N, int Ci, int Co, int H, int H2) {
  int i = blockIdx.x * blockDim.x + threadIdx.x;
  int total = N * Co * H2 * H2;
  if (i >= total) return;
  int ox = i % H2; int t = i / H2;
  int oy = t % H2; t /= H2;
  int o  = t % Co; int n = t / Co;
  float acc = b[o];
  for (int c = 0; c < Ci; ++c) {
    const float* ib = in + ((size_t)n * Ci + c) * H * H;
    const float* wb = w + ((size_t)o * Ci + c) * 4;
    acc += ib[(2 * oy) * H + 2 * ox]         * wb[0];
    acc += ib[(2 * oy) * H + 2 * ox + 1]     * wb[1];
    acc += ib[(2 * oy + 1) * H + 2 * ox]     * wb[2];
    acc += ib[(2 * oy + 1) * H + 2 * ox + 1] * wb[3];
  }
  out[i] = acc;
}

// global average pool over HW
__global__ void k_gap(const float* __restrict__ in, float* __restrict__ out,
                      int N, int C, int HW) {
  int i = blockIdx.x * blockDim.x + threadIdx.x;
  if (i >= N * C) return;
  const float* ib = in + (size_t)i * HW;
  float s = 0.f;
  for (int p = 0; p < HW; ++p) s += ib[p];
  out[i] = s / (float)HW;
}

// Mamba causal depthwise conv (k=4, left pad 3) + SiLU; writes f32 and f16
__global__ void k_conv_silu(const float* __restrict__ xz, const float* __restrict__ cw,
                            const float* __restrict__ cb, float* __restrict__ xr32,
                            half_t* __restrict__ xr16) {
  int i = blockIdx.x * blockDim.x + threadIdx.x;
  if (i >= 4 * 1536) return;
  int d = i % 1536, b = i / 1536;
  for (int t = 0; t < 32; ++t) {
    float acc = cb[d];
    for (int j = 0; j < 4; ++j) {
      int tt = t + j - 3;
      if (tt >= 0) acc += xz[((size_t)(b * 32 + tt)) * 3072 + d] * cw[d * 4 + j];
    }
    float s = acc / (1.f + __expf(-acc));
    xr32[((size_t)(b * 32 + t)) * 1536 + d] = s;
    xr16[((size_t)(b * 32 + t)) * 1536 + d] = (half_t)s;
  }
}

// dt-rank cols (0..47) of proj -> f16 rows padded to K=64 (cols 48..63 = 0)
__global__ void k_slice_dt(const float* __restrict__ proj, half_t* __restrict__ dt16) {
  int i = blockIdx.x * blockDim.x + threadIdx.x;
  if (i >= 128 * 64) return;
  int m = i / 64, j = i % 64;
  dt16[i] = (half_t)(j < 48 ? proj[m * 80 + j] : 0.f);
}

// pad dt_w (1536,48) -> f16 (1536,64), cols 48..63 = 0
__global__ void k_pad_dtw(const float* __restrict__ src, half_t* __restrict__ dst) {
  int i = blockIdx.x * blockDim.x + threadIdx.x;
  if (i >= 1536 * 64) return;
  int n = i / 64, j = i % 64;
  dst[i] = (half_t)(j < 48 ? src[n * 48 + j] : 0.f);
}

// selective scan: thread per (b, d); h state in registers, L=32, D_STATE=16
__global__ void k_scan(const float* __restrict__ xr, const float* __restrict__ xz,
                       const float* __restrict__ proj, const float* __restrict__ dt,
                       const float* __restrict__ Alog, const float* __restrict__ Dp,
                       half_t* __restrict__ y16) {
  int i = blockIdx.x * blockDim.x + threadIdx.x;
  if (i >= 4 * 1536) return;
  int d = i % 1536, b = i / 1536;
  float h[16], a[16];
  for (int s = 0; s < 16; ++s) { h[s] = 0.f; a[s] = -__expf(Alog[d * 16 + s]); }
  float Dv = Dp[d];
  for (int t = 0; t < 32; ++t) {
    int row = b * 32 + t;
    float dtv = dt[(size_t)row * 1536 + d];
    float xv  = xr[(size_t)row * 1536 + d];
    const float* pr = proj + (size_t)row * 80;
    float y = 0.f;
    for (int s = 0; s < 16; ++s) {
      float dA = __expf(dtv * a[s]);
      h[s] = dA * h[s] + dtv * pr[48 + s] * xv;
      y += h[s] * pr[64 + s];
    }
    y += xv * Dv;
    float z = xz[(size_t)row * 3072 + 1536 + d];
    y *= z / (1.f + __expf(-z));
    y16[(size_t)row * 1536 + d] = (half_t)y;
  }
}

// mean over frames: t(4*32,768) -> efm(4,768)
__global__ void k_mean_frames(const float* __restrict__ t, float* __restrict__ efm) {
  int i = blockIdx.x * blockDim.x + threadIdx.x;
  if (i >= 4 * 768) return;
  int b = i / 768, c = i % 768;
  float s = 0.f;
  for (int f = 0; f < 32; ++f) s += t[((size_t)(b * 32 + f)) * 768 + c];
  efm[i] = s / 32.f;
}

__global__ void k_ef1(const float* __restrict__ efm, const float* __restrict__ w1,
                      const float* __restrict__ b1, float* __restrict__ hid) {
  int b = blockIdx.x, j = threadIdx.x;
  float s = b1[j];
  const float* e = efm + b * 768;
  const float* w = w1 + (size_t)j * 768;
  for (int c = 0; c < 768; ++c) s += e[c] * w[c];
  hid[b * 256 + j] = s > 0.f ? s : 0.f;
}

__global__ void k_ef2(const float* __restrict__ hid, const float* __restrict__ w2,
                      const float* __restrict__ b2, float* __restrict__ out) {
  int b = blockIdx.x * blockDim.x + threadIdx.x;
  if (b >= 4) return;
  float s = b2[0];
  const float* h = hid + b * 256;
  for (int j = 0; j < 256; ++j) s += h[j] * w2[j];
  out[b] = 100.f / (1.f + __expf(-s));
}

// repack deconv weights (Cin, Cout, k, k) -> Wt[(o*kk + tap)][Cin] f16
__global__ void k_repack_wt(const float* __restrict__ src, half_t* __restrict__ dst,
                            int Cin, int Cout, int kk) {
  int i = blockIdx.x * blockDim.x + threadIdx.x;
  int total = Cin * Cout * kk;
  if (i >= total) return;
  int c = i % Cin, n = i / Cin;       // n = o*kk + tap
  dst[i] = (half_t)src[(size_t)c * Cout * kk + n];
}

// col2im for stride-s, pad-0 transposed conv:
// out[f,o,yo,xo] = act(b[o] + sum_{valid ky,kx} col[(f*Hin^2 + iy*Hin + ix)*(Cout*k^2) + o*k^2 + ky*k + kx])
__global__ void k_col2im(const half_t* __restrict__ col, const float* __restrict__ b,
                         float* __restrict__ out, int F, int Cout, int Hin, int Hout,
                         int ksz, int stride, int act) {
  int i = blockIdx.x * blockDim.x + threadIdx.x;
  int total = F * Cout * Hout * Hout;
  if (i >= total) return;
  int xo = i % Hout; int t = i / Hout;
  int yo = t % Hout; t /= Hout;
  int o  = t % Cout; int f = t / Cout;
  const int kk = ksz * ksz;
  const size_t rs = (size_t)Cout * kk;
  float acc = b[o];
  for (int ky = 0; ky < ksz; ++ky) {
    int ty = yo - ky;
    if (ty < 0 || (ty % stride)) continue;
    int iy = ty / stride; if (iy >= Hin) continue;
    for (int kx = 0; kx < ksz; ++kx) {
      int tx = xo - kx;
      if (tx < 0 || (tx % stride)) continue;
      int ix = tx / stride; if (ix >= Hin) continue;
      acc += (float)col[((size_t)(f * Hin * Hin + iy * Hin + ix)) * rs
                        + (size_t)o * kk + ky * ksz + kx];
    }
  }
  out[i] = act_f(acc, act);
}

// direct transposed conv (only used for the tiny final 3x3), weight (Cin, Cout, k, k)
__global__ void k_deconv(const float* __restrict__ in, const float* __restrict__ w,
                         const float* __restrict__ b, float* __restrict__ out,
                         int F, int Cin, int Cout, int Hin, int Hout,
                         int ksz, int stride, int pad, int act) {
  int i = blockIdx.x * blockDim.x + threadIdx.x;
  int total = F * Cout * Hout * Hout;
  if (i >= total) return;
  int xo = i % Hout; int t = i / Hout;
  int yo = t % Hout; t /= Hout;
  int o  = t % Cout; int f = t / Cout;
  float acc = b[o];
  const float* ipf = in + (size_t)f * Cin * Hin * Hin;
  for (int ky = 0; ky < ksz; ++ky) {
    int ty = yo + pad - ky;
    if (ty < 0 || (ty % stride)) continue;
    int iy = ty / stride; if (iy >= Hin) continue;
    for (int kx = 0; kx < ksz; ++kx) {
      int tx = xo + pad - kx;
      if (tx < 0 || (tx % stride)) continue;
      int ix = tx / stride; if (ix >= Hin) continue;
      const float* ip = ipf + iy * Hin + ix;
      const float* wp = w + (size_t)o * ksz * ksz + ky * ksz + kx;
      float s = 0.f;
      for (int c = 0; c < Cin; ++c)
        s += ip[(size_t)c * Hin * Hin] * wp[(size_t)c * Cout * ksz * ksz];
      acc += s;
    }
  }
  out[i] = act_f(acc, act);
}

// ---------------- host orchestration ----------------

static void gemm(hipStream_t st, const half_t* A, const half_t* W, const float* bias,
                 float* Cf, half_t* Ch, int M, int N, int K, int act, int bshift) {
  dim3 g(ceil_div(N, 128), ceil_div(M, 128));
  k_gemm_wmma<<<g, 256, 0, st>>>(A, W, bias, Cf, Ch, M, N, K, act, bshift,
                                 Ch != nullptr ? 1 : 0);
}

#define EL_LAUNCH(kern, n, ...) kern<<<ceil_div((int)(n), 256), 256, 0, stream>>>(__VA_ARGS__)

extern "C" void kernel_launch(void* const* d_in, const int* in_sizes, int n_in,
                              void* d_out, int out_size, void* d_ws, size_t ws_size,
                              hipStream_t stream) {
  (void)in_sizes; (void)n_in; (void)out_size; (void)ws_size;
  int idx = 0;
  auto nx = [&]() -> const float* { return (const float*)d_in[idx++]; };

  const float* x = nx();
  const float *stem_w = nx(), *stem_b = nx(), *stem_g = nx(), *stem_bb = nx();

  struct Blk { const float *dw_w, *dw_b, *ln_g, *ln_b, *fc1_w, *fc1_b, *fc2_w, *fc2_b, *gamma; };
  static const int DIMS_[4] = {96, 192, 384, 768};
  static const int DEP_[4]  = {3, 3, 9, 3};
  Blk blk[18]; int nb = 0;
  for (int s = 0; s < 4; ++s)
    for (int d = 0; d < DEP_[s]; ++d) {
      Blk& B = blk[nb++];
      B.dw_w = nx(); B.dw_b = nx(); B.ln_g = nx(); B.ln_b = nx();
      B.fc1_w = nx(); B.fc1_b = nx(); B.fc2_w = nx(); B.fc2_b = nx(); B.gamma = nx();
    }
  struct DownP { const float *g, *b, *w, *bb; } dn[3];
  for (int i = 0; i < 3; ++i) { dn[i].g = nx(); dn[i].b = nx(); dn[i].w = nx(); dn[i].bb = nx(); }
  struct Mb { const float *in_w, *conv_w, *conv_b, *xp_w, *dt_w, *dt_b, *Alog, *Dp, *out_w; } mb[2];
  for (int i = 0; i < 2; ++i) {
    mb[i].in_w = nx(); mb[i].conv_w = nx(); mb[i].conv_b = nx(); mb[i].xp_w = nx();
    mb[i].dt_w = nx(); mb[i].dt_b = nx(); mb[i].Alog = nx(); mb[i].Dp = nx(); mb[i].out_w = nx();
  }
  const float *ef_w1 = nx(), *ef_b1 = nx(), *ef_w2 = nx(), *ef_b2 = nx();
  const float *sg_w[4], *sg_b[4];
  for (int i = 0; i < 4; ++i) { sg_w[i] = nx(); sg_b[i] = nx(); }

  // workspace bump allocator
  char* wp = (char*)d_ws;
  auto alloc = [&](size_t bytes) -> char* {
    char* r = wp; wp += (bytes + 255) & ~(size_t)255; return r;
  };
  const int NF = 128;
  float*  nchwA   = (float*)alloc((size_t)NF * 96 * 28 * 28 * 4);
  float*  nchwB   = (float*)alloc((size_t)NF * 96 * 28 * 28 * 4);
  half_t* rows16a = (half_t*)alloc((size_t)NF * 28 * 28 * 96 * 2);
  half_t* rows16b = (half_t*)alloc((size_t)NF * 28 * 28 * 384 * 2); // also col buffer
  float*  rows32  = (float*)alloc((size_t)NF * 28 * 28 * 96 * 4);
  half_t* wf16    = (half_t*)alloc((size_t)6144 * 768 * 2);
  float*  xz32    = (float*)alloc((size_t)128 * 3072 * 4);
  float*  xr32    = (float*)alloc((size_t)128 * 1536 * 4);
  half_t* xr16    = (half_t*)alloc((size_t)128 * 1536 * 2);
  float*  proj32  = (float*)alloc((size_t)128 * 128 * 4);   // N=80 padded tile
  half_t* dt16    = (half_t*)alloc((size_t)128 * 64 * 2);   // K padded 48->64
  float*  dt32    = (float*)alloc((size_t)128 * 1536 * 4);
  half_t* y16     = (half_t*)alloc((size_t)128 * 1536 * 2);
  float*  tA      = (float*)alloc((size_t)128 * 768 * 4);
  float*  tB      = (float*)alloc((size_t)128 * 768 * 4);
  float*  efm     = (float*)alloc((size_t)4 * 768 * 4);
  float*  hid     = (float*)alloc((size_t)4 * 256 * 4);
  float*  seg2    = (float*)alloc((size_t)NF * 192 * 100 * 4);
  float*  seg3    = (float*)alloc((size_t)NF * 96 * 484 * 4);

  // ---- stem ----
  EL_LAUNCH(k_stem, 128 * 96 * 28 * 28, x, stem_w, stem_b, nchwA);
  EL_LAUNCH(k_ln2d, NF * 784, nchwA, stem_g, stem_bb, NF, 96, 784);

  // ---- ConvNeXt stages ----
  float* cur = nchwA; float* alt = nchwB;
  int H = 28; nb = 0;
  for (int s = 0; s < 4; ++s) {
    int C = DIMS_[s], HW = H * H, M = NF * HW;   // M always a multiple of 128
    for (int d = 0; d < DEP_[s]; ++d) {
      Blk& B = blk[nb++];
      EL_LAUNCH(k_dwconv7, NF * C * HW, cur, B.dw_w, B.dw_b, alt, NF, C, H);
      EL_LAUNCH(k_ln_rows_f16, NF * HW, alt, B.ln_g, B.ln_b, rows16a, NF, C, HW);
      EL_LAUNCH(k_f2h, 4 * C * C, B.fc1_w, wf16, 4 * C * C);
      gemm(stream, rows16a, wf16, B.fc1_b, nullptr, rows16b, M, 4 * C, C, /*gelu*/1, 0);
      EL_LAUNCH(k_f2h, 4 * C * C, B.fc2_w, wf16, 4 * C * C);
      gemm(stream, rows16b, wf16, B.fc2_b, rows32, nullptr, M, C, 4 * C, 0, 0);
      EL_LAUNCH(k_residual, NF * C * HW, cur, rows32, B.gamma, NF, C, HW);
    }
    if (s < 3) {
      int Co = DIMS_[s + 1], H2 = (H - 2) / 2 + 1;
      EL_LAUNCH(k_ln2d, NF * HW, cur, dn[s].g, dn[s].b, NF, C, HW);
      EL_LAUNCH(k_down2, NF * Co * H2 * H2, cur, dn[s].w, dn[s].bb, alt, NF, C, Co, H, H2);
      float* t = cur; cur = alt; alt = t;
      H = H2;
    }
  }

  // ---- global average pool -> (128, 768) ----
  EL_LAUNCH(k_gap, NF * 768, cur, tA, NF, 768, H * H);

  // ---- Mamba x2 ----
  float* tcur = tA; float* tnext = tB;
  for (int mi = 0; mi < 2; ++mi) {
    Mb& m = mb[mi];
    EL_LAUNCH(k_f2h, 128 * 768, tcur, rows16a, 128 * 768);
    EL_LAUNCH(k_f2h, 3072 * 768, m.in_w, wf16, 3072 * 768);
    gemm(stream, rows16a, wf16, nullptr, xz32, nullptr, 128, 3072, 768, 0, 0);
    EL_LAUNCH(k_conv_silu, 4 * 1536, xz32, m.conv_w, m.conv_b, xr32, xr16);
    EL_LAUNCH(k_f2h, 80 * 1536, m.xp_w, wf16, 80 * 1536);
    gemm(stream, xr16, wf16, nullptr, proj32, nullptr, 128, 80, 1536, 0, 0);
    EL_LAUNCH(k_slice_dt, 128 * 64, proj32, dt16);
    EL_LAUNCH(k_pad_dtw, 1536 * 64, m.dt_w, wf16);
    gemm(stream, dt16, wf16, m.dt_b, dt32, nullptr, 128, 1536, 64, /*softplus*/4, 0);
    EL_LAUNCH(k_scan, 4 * 1536, xr32, xz32, proj32, dt32, m.Alog, m.Dp, y16);
    EL_LAUNCH(k_f2h, 768 * 1536, m.out_w, wf16, 768 * 1536);
    gemm(stream, y16, wf16, nullptr, tnext, nullptr, 128, 768, 1536, 0, 0);
    float* t = tcur; tcur = tnext; tnext = t;
  }

  // ---- EF head ----
  EL_LAUNCH(k_mean_frames, 4 * 768, tcur, efm);
  k_ef1<<<4, 256, 0, stream>>>(efm, ef_w1, ef_b1, hid);
  k_ef2<<<1, 32, 0, stream>>>(hid, ef_w2, ef_b2, (float*)d_out);

  // ---- Seg head (all three big deconvs through WMMA) ----
  // deconv0 on 1x1 input == pure GEMM: M=128, K=768, N=384*16, fused bias+relu
  EL_LAUNCH(k_f2h, 128 * 768, tcur, rows16a, 128 * 768);
  EL_LAUNCH(k_repack_wt, 6144 * 768, sg_w[0], wf16, 768, 384, 16);
  gemm(stream, rows16a, wf16, sg_b[0], rows32, nullptr, 128, 6144, 768, /*relu*/3, /*bias>>4*/4);
  // rows32 is now (128, 384, 4, 4) NCHW

  // deconv1: 4->10, as GEMM (M=2048, K=384, N=3072) + col2im(+bias+relu)
  EL_LAUNCH(k_nchw_rows_f16, NF * 384 * 16, rows32, rows16a, NF, 384, 16);
  EL_LAUNCH(k_repack_wt, 384 * 192 * 16, sg_w[1], wf16, 384, 192, 16);
  gemm(stream, rows16a, wf16, nullptr, nullptr, rows16b, NF * 16, 3072, 384, 0, 0);
  EL_LAUNCH(k_col2im, NF * 192 * 100, rows16b, sg_b[1], seg2, NF, 192, 4, 10, 4, 2, 3);

  // deconv2: 10->22, as GEMM (M=12800, K=192, N=1536) + col2im(+bias+relu)
  EL_LAUNCH(k_nchw_rows_f16, NF * 192 * 100, seg2, rows16a, NF, 192, 100);
  EL_LAUNCH(k_repack_wt, 192 * 96 * 16, sg_w[2], wf16, 192, 96, 16);
  gemm(stream, rows16a, wf16, nullptr, nullptr, rows16b, NF * 100, 1536, 192, 0, 0);
  EL_LAUNCH(k_col2im, NF * 96 * 484, rows16b, sg_b[2], seg3, NF, 96, 10, 22, 4, 2, 3);

  // final 3x3 stride-1 pad-1 (tiny) + sigmoid -> d_out
  EL_LAUNCH(k_deconv, NF * 1 * 484, seg3, sg_w[3], sg_b[3], (float*)d_out + 4,
            NF, 96, 1, 22, 22, 3, 1, 1, /*sigmoid*/5);
}
